// GIN_91216515432811
// MI455X (gfx1250) — matrix operand order repaired and checked by reading, not compiled, not run on hardware
//
#include <hip/hip_runtime.h>

// ---------------------------------------------------------------------------
// Types for CDNA5 WMMA (wave32)
// ---------------------------------------------------------------------------
typedef __attribute__((ext_vector_type(16))) __bf16        v16bf;
typedef __attribute__((ext_vector_type(8)))  float         v8f;
typedef __attribute__((ext_vector_type(4)))  unsigned int  u32x4;

struct FragBits { u32x4 lo, hi; };   // 32 bytes == v16bf

__device__ __forceinline__ __bf16 to_bf16(float f) {
    unsigned u = __builtin_bit_cast(unsigned, f);
    unsigned r = u + 0x7FFFu + ((u >> 16) & 1u);   // round-to-nearest-even
    return __builtin_bit_cast(__bf16, (unsigned short)(r >> 16));
}

// ---------------------------------------------------------------------------
// Weight convert + transpose: W[k][n] (row-major, K=128 rows) -> Wt[n][k] bf16.
// Pads columns n in [nCols, nPad) with zeros so the GEMM needs no B-side guard.
// ---------------------------------------------------------------------------
__global__ void __launch_bounds__(256)
convert_wt_kernel(const float* __restrict__ W, __bf16* __restrict__ Wt,
                  int nCols, int nPad) {
    int idx = blockIdx.x * 256 + threadIdx.x;
    int total = nPad * 128;
    if (idx >= total) return;
    int n = idx >> 7;
    int k = idx & 127;
    float v = (n < nCols) ? W[k * nCols + n] : 0.0f;
    Wt[idx] = to_bf16(v);
}

// ---------------------------------------------------------------------------
// Edge scatter: agg[dst] += h[src].  One wave32 per edge, float4 per lane.
// Both h (51.2MB) and agg (51.2MB) are L2-resident on MI455X (192MB L2).
// ---------------------------------------------------------------------------
__global__ void __launch_bounds__(256)
scatter_agg_kernel(const float* __restrict__ H, const int* __restrict__ edge,
                   float* __restrict__ Agg, int nEdges) {
    long long gid = (long long)blockIdx.x * 256 + threadIdx.x;
    int e    = (int)(gid >> 5);
    int lane = (int)(gid & 31);
    if (e >= nEdges) return;
    int s = edge[e];            // edge_index[0][e] : source node
    int d = edge[nEdges + e];   // edge_index[1][e] : destination node
    const float4 v = *(const float4*)(H + (long long)s * 128 + lane * 4);
    float* a = Agg + (long long)d * 128 + lane * 4;
    atomicAdd(a + 0, v.x);
    atomicAdd(a + 1, v.y);
    atomicAdd(a + 2, v.z);
    atomicAdd(a + 3, v.w);
}

// ---------------------------------------------------------------------------
// Fused GIN layer GEMM:
//   A   = (1+eps)*X + Agg            (Agg==nullptr -> A = X)
//   Out = [relu]( A @ W + bias )     (W given transposed, bf16, padded)
//
// NT       = number of 16-wide column tiles (8 for H=128, 3 for padded head)
// RELU     = compile-time ReLU selection (no cndmask in epilogue)
// COLGUARD = compile-time column guard (only the 40-class head needs it)
//
// 256 threads = 8 waves; block covers 128 rows.
//  - Weights (NT*16 x 128 bf16, <=32KB) pulled into LDS with
//    GLOBAL_LOAD_ASYNC_TO_LDS_B128 (ASYNCcnt path), overlapped with the
//    fused fp32 -> bf16 A-tile staging, then s_wait_asynccnt + barrier.
//  - Each wave: one 16-row strip; per column tile preload all 4 B fragments
//    from LDS, then 4 back-to-back v_wmma_f32_16x16x32_bf16.
// ---------------------------------------------------------------------------
template <int NT, bool RELU, bool COLGUARD>
__global__ void __launch_bounds__(256)
gin_gemm_kernel(const float* __restrict__ X, const float* __restrict__ Agg,
                const float* __restrict__ epsPtr,
                const __bf16* __restrict__ Wt,   // [NT*16 x 128] col-major bf16
                const float* __restrict__ bias,
                float* __restrict__ Out,
                int nRows, int nCols) {
    __shared__ alignas(32) __bf16 sA[128 * 128];      // 32 KB staged A tile
    __shared__ alignas(32) __bf16 sW[NT * 16 * 128];  // <=32 KB staged weights

    const int tid  = threadIdx.x;
    const int row0 = blockIdx.x * 128;

    // ---- Kick off async weight copy global -> LDS (tracked by ASYNCcnt).
    // GVS addressing: mem = SGPR64(Wt) + VGPR32(off); vdst = LDS byte address.
    {
        const unsigned wbytes = (unsigned)NT * 4096u;              // NT*16*128*2
        const unsigned swBase = (unsigned)(size_t)(void*)&sW[0];   // LDS = addr[31:0]
        for (unsigned off = (unsigned)tid * 16u; off < wbytes; off += 256u * 16u) {
            unsigned lds = swBase + off;
            asm volatile("global_load_async_to_lds_b128 %0, %1, %2"
                         :: "v"(lds), "v"(off), "s"(Wt)
                         : "memory");
        }
    }

    // ---- Stage A tile (128 x 128) into LDS as bf16, fused (1+eps)*x + agg.
    const float scale = Agg ? (1.0f + *epsPtr) : 1.0f;
    for (int base = tid * 4; base < 128 * 128; base += 256 * 4) {
        int r = base >> 7;
        int c = base & 127;
        int gr = row0 + r;
        float4 xv = make_float4(0.f, 0.f, 0.f, 0.f);
        float4 av = make_float4(0.f, 0.f, 0.f, 0.f);
        if (gr < nRows) {
            xv = *(const float4*)(X + (long long)gr * 128 + c);
            if (Agg) av = *(const float4*)(Agg + (long long)gr * 128 + c);
        }
        sA[base + 0] = to_bf16(scale * xv.x + av.x);
        sA[base + 1] = to_bf16(scale * xv.y + av.y);
        sA[base + 2] = to_bf16(scale * xv.z + av.z);
        sA[base + 3] = to_bf16(scale * xv.w + av.w);
    }

    // Wave's async copies must land, then make LDS visible block-wide.
    asm volatile("s_wait_asynccnt 0x0" ::: "memory");
    __syncthreads();

    // ---- WMMA compute. wave32 layouts (ISA 7.12.2):
    //  A 16x32 bf16: half=L>>4, m=L&15; K = half*16 + 2v  (contiguous along sA row; code's own order, not ISA table)
    //  B 32x16 bf16: n=L&15; K = half*16 + 2v  (contiguous along Wt row; same order as A, not ISA table)
    //  C/D f32: row = half*8 + i, col = L&15
    const int wave = tid >> 5;
    const int lane = tid & 31;
    const int half = lane >> 4;
    const int ln   = lane & 15;
    const int wrow = wave * 16;

    // A fragments for this wave's 16-row strip: loaded once, reused NT times.
    const __bf16* aRow = sA + (wrow + ln) * 128;
    v16bf af[4];
#pragma unroll
    for (int kt = 0; kt < 4; ++kt) {
        FragBits bits;
        bits.lo = *(const u32x4*)(aRow + kt * 32 + half * 16);
        bits.hi = *(const u32x4*)(aRow + kt * 32 + half * 16 + 8);
        af[kt] = __builtin_bit_cast(v16bf, bits);
    }

    // Preload bias for every column tile (no global loads in the hot loop).
    float bv[NT];
#pragma unroll
    for (int nt = 0; nt < NT; ++nt) {
        const int gcol = nt * 16 + ln;
        bv[nt] = (!COLGUARD || gcol < nCols) ? bias[gcol] : 0.0f;
    }

    const bool fullTile = (row0 + 128 <= nRows);   // uniform: no row guards

#pragma unroll
    for (int nt = 0; nt < NT; ++nt) {
        // Preload all 4 B fragments from LDS, then 4 back-to-back WMMAs.
        const __bf16* bRow = sW + (nt * 16 + ln) * 128 + half * 16;
        v16bf bf[4];
#pragma unroll
        for (int kt = 0; kt < 4; ++kt)
            bf[kt] = *(const v16bf*)(bRow + kt * 32);

        v8f acc = {};
#pragma unroll
        for (int kt = 0; kt < 4; ++kt)
            acc = __builtin_amdgcn_wmma_f32_16x16x32_bf16(
                false, af[kt], false, bf[kt], (short)0, acc, false, false);

        // ---- Epilogue: bias + compile-time ReLU, store column.
        const int gcol = nt * 16 + ln;
        if (!COLGUARD || gcol < nCols) {
            if (fullTile) {
                float* colp = Out + (long long)(row0 + wrow + half * 8) * nCols + gcol;
#pragma unroll
                for (int i = 0; i < 8; ++i) {
                    float v = acc[i] + bv[nt];
                    if (RELU) v = fmaxf(v, 0.0f);
                    colp[(long long)i * nCols] = v;
                }
            } else {
#pragma unroll
                for (int i = 0; i < 8; ++i) {
                    int grow = row0 + wrow + half * 8 + i;
                    if (grow < nRows) {
                        float v = acc[i] + bv[nt];
                        if (RELU) v = fmaxf(v, 0.0f);
                        Out[(long long)grow * nCols + gcol] = v;
                    }
                }
            }
        }
    }
}

// ---------------------------------------------------------------------------
// Row-wise log-softmax over C=40 logits.
// ---------------------------------------------------------------------------
__global__ void __launch_bounds__(256)
log_softmax_kernel(const float* __restrict__ logits, float* __restrict__ out,
                   int nRows, int C) {
    int r = blockIdx.x * 256 + threadIdx.x;
    if (r >= nRows) return;
    const float* p = logits + (long long)r * C;
    float mx = -3.4e38f;
    for (int c = 0; c < C; ++c) mx = fmaxf(mx, p[c]);
    float s = 0.0f;
    for (int c = 0; c < C; ++c) s += __expf(p[c] - mx);
    const float lse = __logf(s) + mx;
    float* o = out + (long long)r * C;
    for (int c = 0; c < C; ++c) o[c] = p[c] - lse;
}

// ---------------------------------------------------------------------------
// Host-side orchestration (graph-capture safe: only async memset + launches)
// ---------------------------------------------------------------------------
extern "C" void kernel_launch(void* const* d_in, const int* in_sizes, int n_in,
                              void* d_out, int out_size, void* d_ws, size_t ws_size,
                              hipStream_t stream) {
    const float* x    = (const float*)d_in[0];
    const int*   edge = (const int*)d_in[1];
    const float* eps1 = (const float*)d_in[2];
    const float* W1   = (const float*)d_in[3];
    const float* b1   = (const float*)d_in[4];
    const float* eps2 = (const float*)d_in[5];
    const float* W2   = (const float*)d_in[6];
    const float* b2   = (const float*)d_in[7];
    const float* eps3 = (const float*)d_in[8];
    const float* W3   = (const float*)d_in[9];
    const float* b3   = (const float*)d_in[10];
    const float* Wl1  = (const float*)d_in[11];
    const float* bl1  = (const float*)d_in[12];
    const float* Wl2  = (const float*)d_in[13];
    const float* bl2  = (const float*)d_in[14];
    float* out = (float*)d_out;

    const int N = 100000, E = 1600000, H = 128, C = 40, CPAD = 48;
    const size_t NHB = (size_t)N * H * sizeof(float);   // 51.2 MB

    char* ws = (char*)d_ws;
    float*  hA   = (float*)(ws);
    float*  hB   = (float*)(ws + NHB);
    float*  agg  = (float*)(ws + 2 * NHB);
    __bf16* wt1  = (__bf16*)(ws + 3 * NHB);
    __bf16* wt2  = wt1  + 128 * 128;
    __bf16* wt3  = wt2  + 128 * 128;
    __bf16* wtl1 = wt3  + 128 * 128;
    __bf16* wtl2 = wtl1 + 128 * 128;                    // 48 x 128 (padded)

    const dim3 blk(256);
    const int cvtBlocks  = (128 * 128 + 255) / 256;
    const int cvtBlocksC = (CPAD * 128 + 255) / 256;
    const int gemmBlocks = (N + 127) / 128;
    const int scatBlocks = (int)(((long long)E * 32 + 255) / 256);
    const int smBlocks   = (N + 255) / 256;

    // Weight convert + transpose to bf16 (tiny; all L2-resident afterwards)
    convert_wt_kernel<<<cvtBlocks,  blk, 0, stream>>>(W1,  wt1,  H, H);
    convert_wt_kernel<<<cvtBlocks,  blk, 0, stream>>>(W2,  wt2,  H, H);
    convert_wt_kernel<<<cvtBlocks,  blk, 0, stream>>>(W3,  wt3,  H, H);
    convert_wt_kernel<<<cvtBlocks,  blk, 0, stream>>>(Wl1, wtl1, H, H);
    convert_wt_kernel<<<cvtBlocksC, blk, 0, stream>>>(Wl2, wtl2, C, CPAD);

    // ---- GIN layer 1
    hipMemsetAsync(agg, 0, NHB, stream);
    scatter_agg_kernel<<<scatBlocks, blk, 0, stream>>>(x, edge, agg, E);
    gin_gemm_kernel<8, true, false><<<gemmBlocks, blk, 0, stream>>>(
        x, agg, eps1, wt1, b1, hA, N, H);

    // ---- GIN layer 2
    hipMemsetAsync(agg, 0, NHB, stream);
    scatter_agg_kernel<<<scatBlocks, blk, 0, stream>>>(hA, edge, agg, E);
    gin_gemm_kernel<8, true, false><<<gemmBlocks, blk, 0, stream>>>(
        hA, agg, eps2, wt2, b2, hB, N, H);

    // ---- GIN layer 3
    hipMemsetAsync(agg, 0, NHB, stream);
    scatter_agg_kernel<<<scatBlocks, blk, 0, stream>>>(hB, edge, agg, E);
    gin_gemm_kernel<8, true, false><<<gemmBlocks, blk, 0, stream>>>(
        hB, agg, eps3, wt3, b3, hA, N, H);

    // ---- Linear + ReLU
    gin_gemm_kernel<8, true, false><<<gemmBlocks, blk, 0, stream>>>(
        hA, nullptr, nullptr, wtl1, bl1, hB, N, H);

    // ---- Head (128 -> 40), logits into agg scratch (free by now)
    gin_gemm_kernel<3, false, true><<<gemmBlocks, blk, 0, stream>>>(
        hB, nullptr, nullptr, wtl2, bl2, agg, N, C);

    // ---- log_softmax
    log_softmax_kernel<<<smBlocks, blk, 0, stream>>>(agg, out, N, C);
}
